// EncoderClassifier_36069135352208
// MI455X (gfx1250) — matrix operand order repaired and checked by reading, not compiled, not run on hardware
//
#include <hip/hip_runtime.h>
#include <math.h>
#include <stdint.h>

// ---------------- CDNA5 WMMA types ----------------
typedef __attribute__((ext_vector_type(16))) __bf16 v16bf;
typedef __attribute__((ext_vector_type(8)))  float  v8f;
typedef unsigned int u32x4 __attribute__((ext_vector_type(4)));
typedef int          i32x8 __attribute__((ext_vector_type(8)));
typedef int          i32x4 __attribute__((ext_vector_type(4)));

typedef unsigned short bf16b;   // bf16 bit-pattern storage (host-safe)

#define BQ    8
#define LQ    1024
#define DM    768
#define HH    12
#define DKH   64
#define FF    3072
#define NLAY  6
#define MROWS (BQ*LQ)           // 8192

__device__ __forceinline__ bf16b f2bf(float f) {
  unsigned u = __builtin_bit_cast(unsigned, f);
  return (bf16b)((u + 0x7FFFu + ((u >> 16) & 1u)) >> 16);   // RNE
}

union Frag {
  v16bf v;
  uint4 q[2];
};

__device__ __forceinline__ v8f wmma_bf16(const Frag& a, const Frag& b, v8f c) {
  return __builtin_amdgcn_wmma_f32_16x16x32_bf16(
      /*neg_a=*/false, a.v, /*neg_b=*/false, b.v,
      /*c_mod=*/(short)0, c, /*reuse_a=*/false, /*reuse_b=*/false);
}

// ---------------- Tensor Data Mover: 16-row x 64-elem bf16 tile -> LDS ----------------
// D# per CDNA5 ISA. 2-D tensor, data_size=2B, count=1, type=2.
// LDS pad: rows are 64 elems (128B = 32 DWORDs) padded by 16B (4 DWORDs) to a 72-halfword
// stride => pad_interval code 4 (32 DWORDs), pad_amount code 3 (4 DWORDs).
// Toolchain here is the 6-arg clang-23 builtin:
//   (uint32x4 g0, int32x8 g1, int32x4 g2, int32x4 g3, int32x8 g4, i32 cpol)
__device__ __forceinline__ void tdm_load_tile16(
    unsigned lds_addr, const bf16b* gptr, int K, int tensor_rows)
{
  union { unsigned long long q[2]; u32x4 v; } g0;
  union { unsigned long long q[4]; i32x8 v; } g1;
  unsigned long long ga = (unsigned long long)(uintptr_t)gptr;
  g0.q[0] = 1ull                                        // count = 1 (valid descriptor)
          | ((unsigned long long)lds_addr << 32);       // lds_addr [63:32]
  g0.q[1] = (ga & 0x01FFFFFFFFFFFFFFull)                // global_addr [120:64]
          | (2ull << 62);                               // type = 2 ("image") [127:126]
  unsigned long long flags =
      (1ull << 16)                                      // data_size = 2 bytes
    | (1ull << 20)                                      // pad_enable
    | (4ull << 22)                                      // pad_interval = 32 DWORDs
    | (3ull << 25);                                     // pad_amount = 4 DWORDs
  g1.q[0] = flags
          | (((unsigned long long)(unsigned)K & 0xFFFFull) << 48);   // tensor_dim0[15:0]
  g1.q[1] = ((unsigned long long)(unsigned)K >> 16)                  // tensor_dim0[31:16]
          | ((unsigned long long)(unsigned)tensor_rows << 16)        // tensor_dim1
          | (64ull << 48);                                           // tile_dim0 = 64
  g1.q[2] = 16ull                                                    // tile_dim1 = 16
          | (((unsigned long long)(unsigned)K) << 32);               // dim0_stride[31:0]
  g1.q[3] = 0;                                                       // stride hi, dim1_stride
  i32x4 z4 = {};
  i32x8 z8 = {};
  __builtin_amdgcn_tensor_load_to_lds(g0.v, g1.v, z4, z4, z8, 0);
}

// ---------------- bf16 WMMA GEMM: out[M,N] = A[M,K] * Bt[N,K]^T + bias ----------------
// Block tile 128x128, BK=64. 8 waves (256 thr), wave grid 2(M) x 4(N), 64x32 per wave.
// Tiles staged by the TDM (tensor_load_to_lds), one 16-row slice per wave.
#define GBM 128
#define GBN 128
#define GBK 64
#define SAS2 72   // LDS row stride (halfwords): 64 + 8 pad -> 144B, 16B aligned

__global__ __launch_bounds__(256) void gemm_bf16(
    const bf16b* __restrict__ A, const bf16b* __restrict__ Bt,
    const float* __restrict__ bias, void* __restrict__ outp,
    int M, int N, int K, int outBf16, int relu)
{
  __shared__ __align__(16) bf16b sA[GBM * SAS2];
  __shared__ __align__(16) bf16b sB[GBN * SAS2];
  const int tid  = threadIdx.x;
  const int lane = tid & 31;
  const int wv   = tid >> 5;
  const int wm   = wv & 1, wn = wv >> 1;     // 2 x 4 wave grid
  const int lm   = lane & 15, hl = lane >> 4;
  const int m0 = blockIdx.y * GBM, n0 = blockIdx.x * GBN;

  const unsigned ldsA = (unsigned)(uintptr_t)&sA[0] + (unsigned)(wv * 16 * SAS2 * 2);
  const unsigned ldsB = (unsigned)(uintptr_t)&sB[0] + (unsigned)(wv * 16 * SAS2 * 2);

  v8f acc[4][2] = {};

  for (int k0 = 0; k0 < K; k0 += GBK) {
    // each wave DMAs its 16-row slice of the A and B tiles into LDS
    tdm_load_tile16(ldsA, A  + (size_t)(m0 + wv * 16) * K + k0, K, M);
    tdm_load_tile16(ldsB, Bt + (size_t)(n0 + wv * 16) * K + k0, K, N);
    if (k0 + GBK < K) {   // hint next K-tile -> global_prefetch_b8
      int e = tid * 8; int r = e >> 6, c = e & 63;
      __builtin_prefetch(&A [(size_t)(m0 + r) * K + k0 + GBK + c], 0, 0);
      __builtin_prefetch(&Bt[(size_t)(n0 + r) * K + k0 + GBK + c], 0, 0);
    }
    __builtin_amdgcn_s_wait_tensorcnt(0);
    __syncthreads();

#pragma unroll
    for (int kc = 0; kc < 2; ++kc) {          // two 32-wide WMMA k-steps per stage
      Frag a[4], b[2];
#pragma unroll
      for (int mt = 0; mt < 4; ++mt) {        // A frag: K 0-7/16-23 (hl=0), 8-15/24-31 (hl=1)
        int m = wm * 64 + mt * 16 + lm;
        a[mt].q[0] = *(const uint4*)&sA[m * SAS2 + kc * 32 + hl * 8];
        a[mt].q[1] = *(const uint4*)&sA[m * SAS2 + kc * 32 + 16 + hl * 8];
      }
#pragma unroll
      for (int nt = 0; nt < 2; ++nt) {        // B frag: K 0-7/16-23 (hl=0), 8-15/24-31 (hl=1)
        int n = wn * 32 + nt * 16 + lm;
        b[nt].q[0] = *(const uint4*)&sB[n * SAS2 + kc * 32 + hl * 8];
        b[nt].q[1] = *(const uint4*)&sB[n * SAS2 + kc * 32 + 16 + hl * 8];
      }
#pragma unroll
      for (int mt = 0; mt < 4; ++mt)
#pragma unroll
        for (int nt = 0; nt < 2; ++nt)
          acc[mt][nt] = wmma_bf16(a[mt], b[nt], acc[mt][nt]);
    }
    __syncthreads();
  }

  // epilogue: C layout lane n = lane%16, row m = r + 8*(lane/16)
#pragma unroll
  for (int mt = 0; mt < 4; ++mt)
    for (int nt = 0; nt < 2; ++nt) {
      int n = n0 + wn * 32 + nt * 16 + lm;
      float bvv = bias ? bias[n] : 0.0f;
#pragma unroll
      for (int r = 0; r < 8; ++r) {
        int m = m0 + wm * 64 + mt * 16 + hl * 8 + r;
        float v = acc[mt][nt][r] + bvv;
        if (relu) v = fmaxf(v, 0.0f);
        if (outBf16) ((bf16b*)outp)[(size_t)m * N + n] = f2bf(v);
        else         ((float*)outp)[(size_t)m * N + n] = v;
      }
    }
}

// ---------------- Flash attention (bf16 WMMA, online softmax) ----------------
// grid = (L/64, H, B); 128 threads = 4 waves; each wave owns 16 query rows.
__global__ __launch_bounds__(128) void flash_attn(
    const bf16b* __restrict__ Q, const bf16b* __restrict__ Kb,
    const bf16b* __restrict__ Vb, const int* __restrict__ mask,
    bf16b* __restrict__ O)
{
  __shared__ __align__(16) bf16b sK[32 * 72];       // 32 keys x 64 dk (pad to 72)
  __shared__ __align__(16) bf16b sV[64 * 40];       // transposed: 64 dk x 32 keys (pad 40)
  __shared__ __align__(16) bf16b sP[4][16 * 40];    // per-wave P scratch 16x32 (pad 40)
  const int b = blockIdx.z, h = blockIdx.y, qt = blockIdx.x;
  const int tid = threadIdx.x, lane = tid & 31, wv = tid >> 5;
  const int lm = lane & 15, hl = lane >> 4;
  const int qrow = qt * 64 + wv * 16;

  Frag aq[2];                                       // Q tile 16x64 as two A-frags
  {
    const bf16b* qp = Q + (size_t)(b * LQ + qrow + lm) * DM + h * DKH;
#pragma unroll
    for (int ch = 0; ch < 2; ++ch) {
      aq[ch].q[0] = *(const uint4*)(qp + ch * 32 + hl * 8);
      aq[ch].q[1] = *(const uint4*)(qp + ch * 32 + 16 + hl * 8);
    }
  }

  v8f o[4] = {};
  float mrow[8], lrow[8];
#pragma unroll
  for (int r = 0; r < 8; ++r) { mrow[r] = -3.0e38f; lrow[r] = 0.0f; }
  const float scale = 0.125f;                       // 1/sqrt(64)

  for (int kc = 0; kc < LQ; kc += 32) {
    // stage K tile 32x64 row-major
#pragma unroll
    for (int i = 0; i < 2; ++i) {
      int e = (tid + i * 128) * 8;
      int r = e >> 6, c = e & 63;
      *(uint4*)&sK[r * 72 + c] = *(const uint4*)&Kb[(size_t)(b * LQ + kc + r) * DM + h * DKH + c];
    }
    // stage V tile transposed: sV[dk][key]
#pragma unroll
    for (int i = 0; i < 2; ++i) {
      int e = (tid + i * 128) * 8;
      int key = e >> 6, dk = e & 63;
      uint4 raw = *(const uint4*)&Vb[(size_t)(b * LQ + kc + key) * DM + h * DKH + dk];
      bf16b tv[8];
      *(uint4*)tv = raw;
#pragma unroll
      for (int j = 0; j < 8; ++j) sV[(dk + j) * 40 + key] = tv[j];
    }
    __syncthreads();

    // S = Q * K^T : 16x32 scores = 2 N-tiles x 2 K-chunks
    v8f s[2] = {};
#pragma unroll
    for (int nt = 0; nt < 2; ++nt) {
      int kr = nt * 16 + lm;
#pragma unroll
      for (int ch = 0; ch < 2; ++ch) {
        Frag bk;
        bk.q[0] = *(const uint4*)&sK[kr * 72 + ch * 32 + hl * 8];
        bk.q[1] = *(const uint4*)&sK[kr * 72 + ch * 32 + 16 + hl * 8];
        s[nt] = wmma_bf16(aq[ch], bk, s[nt]);
      }
    }
    float neg[2];
#pragma unroll
    for (int nt = 0; nt < 2; ++nt) {
      int key = kc + nt * 16 + lm;
      neg[nt] = (mask[b * LQ + key] == 0) ? -1e9f : 0.0f;
    }
    // online softmax per row (row m = r + 8*hl; 16-lane xor reductions stay in-half)
#pragma unroll
    for (int r = 0; r < 8; ++r) {
      float s0 = s[0][r] * scale + neg[0];
      float s1 = s[1][r] * scale + neg[1];
      float v = fmaxf(s0, s1);
#pragma unroll
      for (int off = 1; off < 16; off <<= 1) v = fmaxf(v, __shfl_xor(v, off, 32));
      float mnew  = fmaxf(mrow[r], v);
      float alpha = __expf(mrow[r] - mnew);
      float p0 = __expf(s0 - mnew);
      float p1 = __expf(s1 - mnew);
      float ps = p0 + p1;
#pragma unroll
      for (int off = 1; off < 16; off <<= 1) ps += __shfl_xor(ps, off, 32);
      lrow[r] = lrow[r] * alpha + ps;
      mrow[r] = mnew;
#pragma unroll
      for (int dt = 0; dt < 4; ++dt) o[dt][r] *= alpha;
      sP[wv][(r + 8 * hl) * 40 + lm]      = f2bf(p0);
      sP[wv][(r + 8 * hl) * 40 + 16 + lm] = f2bf(p1);
    }
    // P (A-frag) * V  -> accumulate O (16x64)
    Frag ap;
    ap.q[0] = *(const uint4*)&sP[wv][lm * 40 + hl * 8];
    ap.q[1] = *(const uint4*)&sP[wv][lm * 40 + 16 + hl * 8];
#pragma unroll
    for (int dt = 0; dt < 4; ++dt) {
      Frag bv;
      int dr = dt * 16 + lm;
      bv.q[0] = *(const uint4*)&sV[dr * 40 + hl * 8];
      bv.q[1] = *(const uint4*)&sV[dr * 40 + 16 + hl * 8];
      o[dt] = wmma_bf16(ap, bv, o[dt]);
    }
    __syncthreads();
  }

#pragma unroll
  for (int dt = 0; dt < 4; ++dt)
#pragma unroll
    for (int r = 0; r < 8; ++r) {
      float v = o[dt][r] / lrow[r];
      O[(size_t)(b * LQ + qrow + hl * 8 + r) * DM + h * DKH + dt * 16 + lm] = f2bf(v);
    }
}

// ---------------- embedding + sinusoidal PE ----------------
__global__ void embed_pe(const int* __restrict__ ids, const float* __restrict__ emb,
                         float* __restrict__ x, bf16b* __restrict__ xb)
{
  size_t idx = (size_t)blockIdx.x * 256 + threadIdx.x;
  int d = (int)(idx % DM);
  size_t tok = idx / DM;
  int pos = (int)(tok % LQ);
  int id = ids[tok];
  float freq = __expf(-(float)(d & ~1) * (9.2103403719761836f / (float)DM));
  float ang  = (float)pos * freq;
  float pe   = (d & 1) ? __cosf(ang) : __sinf(ang);
  float v = emb[(size_t)id * DM + d] + pe;
  x[idx]  = v;
  xb[idx] = f2bf(v);
}

// ---------------- weight convert fp32 [K,N] -> bf16 transposed [N,K] ----------------
__global__ void conv_w(const float* __restrict__ w, bf16b* __restrict__ wt, int K, int N)
{
  size_t idx = (size_t)blockIdx.x * 256 + threadIdx.x;
  if (idx >= (size_t)K * N) return;
  int n = (int)(idx % N);
  int k = (int)(idx / N);
  wt[(size_t)n * K + k] = f2bf(w[idx]);
}

// ---------------- fused residual + LayerNorm (one block per row) ----------------
__device__ __forceinline__ float block_sum(float v, float* red)
{
#pragma unroll
  for (int off = 16; off > 0; off >>= 1) v += __shfl_xor(v, off, 32);
  if ((threadIdx.x & 31) == 0) red[threadIdx.x >> 5] = v;
  __syncthreads();
  float s = 0.0f;
#pragma unroll
  for (int i = 0; i < 8; ++i) s += red[i];
  __syncthreads();
  return s;
}

__global__ __launch_bounds__(256) void residual_ln(
    float* __restrict__ x, const float* __restrict__ t,
    const float* __restrict__ g, const float* __restrict__ bb,
    bf16b* __restrict__ xb)
{
  __shared__ float red[8];
  size_t row = blockIdx.x;
  float v[3];
  float s = 0.0f;
#pragma unroll
  for (int i = 0; i < 3; ++i) {
    int c = threadIdx.x + i * 256;
    v[i] = x[row * DM + c] + t[row * DM + c];
    s += v[i];
  }
  float mu = block_sum(s, red) * (1.0f / (float)DM);
  float q = 0.0f;
#pragma unroll
  for (int i = 0; i < 3; ++i) { float d = v[i] - mu; q += d * d; }
  float rstd = rsqrtf(block_sum(q, red) * (1.0f / (float)DM) + 1e-5f);
#pragma unroll
  for (int i = 0; i < 3; ++i) {
    int c = threadIdx.x + i * 256;
    float o = (v[i] - mu) * rstd * g[c] + bb[c];
    x[row * DM + c]  = o;
    xb[row * DM + c] = f2bf(o);
  }
}

// ---------------- classifier head (8x10, trivial) ----------------
__global__ void classifier_k(const float* __restrict__ x, const float* __restrict__ w,
                             const float* __restrict__ bb, float* __restrict__ out)
{
  int t = threadIdx.x;
  if (t >= BQ * 10) return;
  int b = t / 10, c = t % 10;
  const float* xr = x + (size_t)b * LQ * DM;   // row (b, 0)
  float s = bb[c];
  for (int d = 0; d < DM; ++d) s += xr[d] * w[d * 10 + c];
  out[t] = s;
}

// ---------------- launcher ----------------
extern "C" void kernel_launch(void* const* d_in, const int* in_sizes, int n_in,
                              void* d_out, int out_size, void* d_ws, size_t ws_size,
                              hipStream_t stream)
{
  (void)in_sizes; (void)n_in; (void)out_size; (void)ws_size;
  const int*   ids  = (const int*)d_in[0];
  const int*   mask = (const int*)d_in[1];
  const float* emb  = (const float*)d_in[2];
  const float* wq   = (const float*)d_in[3];
  const float* bq   = (const float*)d_in[4];
  const float* wk   = (const float*)d_in[5];
  const float* bk   = (const float*)d_in[6];
  const float* wv   = (const float*)d_in[7];
  const float* bv   = (const float*)d_in[8];
  const float* wo   = (const float*)d_in[9];
  const float* bo   = (const float*)d_in[10];
  const float* ln1g = (const float*)d_in[11];
  const float* ln1b = (const float*)d_in[12];
  const float* w1   = (const float*)d_in[13];
  const float* b1   = (const float*)d_in[14];
  const float* w2   = (const float*)d_in[15];
  const float* b2   = (const float*)d_in[16];
  const float* ln2g = (const float*)d_in[17];
  const float* ln2b = (const float*)d_in[18];
  const float* clsw = (const float*)d_in[19];
  const float* clsb = (const float*)d_in[20];
  float* out = (float*)d_out;

  char* ws = (char*)d_ws;
  size_t off = 0;
  auto alloc = [&](size_t bytes) -> void* {
    void* p = ws + off;
    off += (bytes + 255) & ~(size_t)255;
    return p;
  };
  float* x    = (float*)alloc((size_t)MROWS * DM * 4);
  float* tmp  = (float*)alloc((size_t)MROWS * DM * 4);
  bf16b* xb   = (bf16b*)alloc((size_t)MROWS * DM * 2);
  bf16b* qb   = (bf16b*)alloc((size_t)MROWS * DM * 2);
  bf16b* kb2  = (bf16b*)alloc((size_t)MROWS * DM * 2);
  bf16b* vb2  = (bf16b*)alloc((size_t)MROWS * DM * 2);
  bf16b* ab   = (bf16b*)alloc((size_t)MROWS * DM * 2);
  bf16b* hb   = (bf16b*)alloc((size_t)MROWS * FF * 2);
  bf16b* wtq  = (bf16b*)alloc((size_t)DM * DM * 2);
  bf16b* wtk  = (bf16b*)alloc((size_t)DM * DM * 2);
  bf16b* wtv  = (bf16b*)alloc((size_t)DM * DM * 2);
  bf16b* wto  = (bf16b*)alloc((size_t)DM * DM * 2);
  bf16b* wt1  = (bf16b*)alloc((size_t)DM * FF * 2);
  bf16b* wt2  = (bf16b*)alloc((size_t)FF * DM * 2);

  embed_pe<<<(MROWS * DM) / 256, 256, 0, stream>>>(ids, emb, x, xb);

  dim3 gD(DM / GBN, MROWS / GBM);   // (6, 64)
  dim3 gF(FF / GBN, MROWS / GBM);   // (24, 64)
  const int cwDD = (DM * DM + 255) / 256;
  const int cwDF = (DM * FF + 255) / 256;

  for (int l = 0; l < NLAY; ++l) {
    const size_t oDD = (size_t)l * DM * DM;
    const size_t oDF = (size_t)l * DM * FF;
    const size_t oD  = (size_t)l * DM;
    const size_t oF  = (size_t)l * FF;

    conv_w<<<cwDD, 256, 0, stream>>>(wq + oDD, wtq, DM, DM);
    conv_w<<<cwDD, 256, 0, stream>>>(wk + oDD, wtk, DM, DM);
    conv_w<<<cwDD, 256, 0, stream>>>(wv + oDD, wtv, DM, DM);
    conv_w<<<cwDD, 256, 0, stream>>>(wo + oDD, wto, DM, DM);
    conv_w<<<cwDF, 256, 0, stream>>>(w1 + oDF, wt1, DM, FF);
    conv_w<<<cwDF, 256, 0, stream>>>(w2 + oDF, wt2, FF, DM);

    gemm_bf16<<<gD, 256, 0, stream>>>(xb, wtq, bq + oD, qb,  MROWS, DM, DM, 1, 0);
    gemm_bf16<<<gD, 256, 0, stream>>>(xb, wtk, bk + oD, kb2, MROWS, DM, DM, 1, 0);
    gemm_bf16<<<gD, 256, 0, stream>>>(xb, wtv, bv + oD, vb2, MROWS, DM, DM, 1, 0);

    flash_attn<<<dim3(LQ / 64, HH, BQ), 128, 0, stream>>>(qb, kb2, vb2, mask, ab);

    gemm_bf16<<<gD, 256, 0, stream>>>(ab, wto, bo + oD, tmp, MROWS, DM, DM, 0, 0);
    residual_ln<<<MROWS, 256, 0, stream>>>(x, tmp, ln1g + oD, ln1b + oD, xb);

    gemm_bf16<<<gF, 256, 0, stream>>>(xb, wt1, b1 + oF, hb,  MROWS, FF, DM, 1, 1);
    gemm_bf16<<<gD, 256, 0, stream>>>(hb, wt2, b2 + oD, tmp, MROWS, DM, FF, 0, 0);
    residual_ln<<<MROWS, 256, 0, stream>>>(x, tmp, ln2g + oD, ln2b + oD, xb);
  }

  classifier_k<<<1, 128, 0, stream>>>(x, clsw, clsb, out);
}
